// AttFlat_403726926050
// MI455X (gfx1250) — matrix operand order repaired and checked by reading, not compiled, not run on hardware
//
#include <hip/hip_runtime.h>
#include <hip/hip_bf16.h>

typedef __attribute__((ext_vector_type(16))) _Float16 v16h;
typedef __attribute__((ext_vector_type(8)))  _Float16 v8h;
typedef __attribute__((ext_vector_type(4)))  _Float16 v4h;
typedef __attribute__((ext_vector_type(8)))  float    v8f;

#define TWO_PI_F 6.2831853071795864f

__device__ __forceinline__ float lin14(int i) { return (1.0f + (float)i * (12.0f / 13.0f)) / 14.0f; }
__device__ __forceinline__ float lin10(int i) { return (float)i * (1.0f / 9.0f); }

// ---------------------------------------------------------------------------
// Constant-matrix pipeline: F (100x196) -> AUG=[A|I] (100x200) -> inv -> G (196x100)
// ---------------------------------------------------------------------------
__global__ void k_F(float* __restrict__ F) {
    int i = blockIdx.x * 256 + threadIdx.x;
    if (i >= 100 * 196) return;
    int n = i / 196, l = i % 196;
    float px = lin14(l / 14), py = lin14(l % 14);
    float mx = lin10(n / 10), my = lin10(n % 10);
    float d2 = (px - mx) * (px - mx) + (py - my) * (py - my);
    F[i] = expf(-500.0f * d2) * 159.154943091895336f; // 1/(2*pi*0.001)
}

__global__ void k_A(const float* __restrict__ F, float* __restrict__ aug) {
    int i = blockIdx.x * 256 + threadIdx.x;
    if (i >= 100 * 200) return;
    int r = i / 200, c = i % 200;
    if (c < 100) {
        float s = 0.0f;
        for (int l = 0; l < 196; ++l) s += F[r * 196 + l] * F[c * 196 + l];
        if (r == c) s += 0.5f;
        aug[i] = s;
    } else {
        aug[i] = (c - 100 == r) ? 1.0f : 0.0f;
    }
}

// SPD matrix: Gauss-Jordan without pivoting, single block.
__global__ void k_inv(float* __restrict__ aug) {
    int t = threadIdx.x;
    __shared__ float pivinv;
    for (int k = 0; k < 100; ++k) {
        if (t == 0) pivinv = 1.0f / aug[k * 200 + k];
        __syncthreads();
        for (int j = t; j < 200; j += 256) aug[k * 200 + j] *= pivinv;
        __syncthreads();
        if (t < 100 && t != k) {
            float f = aug[t * 200 + k];
            for (int j = 0; j < 200; ++j) aug[t * 200 + j] -= f * aug[k * 200 + j];
        }
        __syncthreads();
    }
}

__global__ void k_G(const float* __restrict__ F, const float* __restrict__ aug,
                    float* __restrict__ G) {
    int i = blockIdx.x * 256 + threadIdx.x;
    if (i >= 196 * 100) return;
    int l = i / 100, n = i % 100;
    float s = 0.0f;
    for (int k = 0; k < 100; ++k) s += F[k * 196 + l] * aug[k * 200 + 100 + n];
    G[i] = s;
}

// ---------------------------------------------------------------------------
// Convert+transpose fp32 [R][C] -> f16 [C][R] (so B fragments are contiguous in K)
// ---------------------------------------------------------------------------
__global__ void k_cvtT(const float* __restrict__ src, _Float16* __restrict__ dst,
                       int R, int C) {
    long i = (long)blockIdx.x * 256 + threadIdx.x;
    if (i >= (long)R * C) return;
    int r = (int)(i / C), c = (int)(i % C);
    dst[(long)c * R + r] = (_Float16)src[i];
}

// ---------------------------------------------------------------------------
// GEMM-1 fused: logits[row] = relu(x_row @ W1 + b1) . W2[:,0] + b2[0]  (+mask)
// 16-row tile per block. x tile staged to LDS with GLOBAL_LOAD_ASYNC_TO_LDS_B128
// (ASYNCcnt path), converted fp32->f16 in LDS, then 8 waves x 4 concurrent
// column-block accumulators sharing one A fragment per WMMA k-step.
// Dynamic LDS: 64KB fp32 staging + 16*1040*2 f16 tile = 98816 B (WGP has 320KB).
// ---------------------------------------------------------------------------
#define XT_STRIDE 1040
#define LDS_F16_OFF 65536
#define K_LOGITS_LDS (LDS_F16_OFF + 16 * XT_STRIDE * 2)

__global__ __launch_bounds__(256) void k_logits(
    const float* __restrict__ x, const _Float16* __restrict__ W1t,
    const float* __restrict__ b1, const float* __restrict__ W2,
    const float* __restrict__ b2, const unsigned char* __restrict__ mask,
    float* __restrict__ logits) {
    extern __shared__ __align__(16) char smem[];
    __shared__ float rowacc[16];
    const int tid = threadIdx.x;
    const int lane = tid & 31;
    const int wave = tid >> 5;
    const long row0 = (long)blockIdx.x * 16;

    // -- async-copy 16x1024 fp32 tile (64KB) into LDS, 16B per lane per op --
    {
        const unsigned ldsbase = (unsigned)(uintptr_t)smem;
        const float* xsrc = x + row0 * 1024;
        for (int i = tid; i < 4096; i += 256) {
            unsigned laddr = ldsbase + (unsigned)i * 16u;
            unsigned long long ga = (unsigned long long)(const void*)(xsrc + i * 4);
            asm volatile("global_load_async_to_lds_b128 %0, %1, off"
                         :: "v"(laddr), "v"(ga) : "memory");
        }
        asm volatile("s_wait_asynccnt 0" ::: "memory");
    }
    if (tid < 16) rowacc[tid] = 0.0f;
    __syncthreads();

    // -- in-LDS fp32 -> f16 conversion pass (padded stride) --
    {
        const float* xf = (const float*)smem;
        _Float16* xh = (_Float16*)(smem + LDS_F16_OFF);
        for (int i = tid; i < 4096; i += 256) {
            int m = i >> 8;             // 256 float4 per 1024-wide row
            int k4 = (i & 255) * 4;
            const float* s4 = xf + m * 1024 + k4;
            v4h hv = { (_Float16)s4[0], (_Float16)s4[1], (_Float16)s4[2], (_Float16)s4[3] };
            *(v4h*)(xh + m * XT_STRIDE + k4) = hv;
        }
    }
    __syncthreads();

    const _Float16* xt = (const _Float16*)(smem + LDS_F16_OFF);
    const int n_lane = lane & 15;
    const int a_koff = (lane >> 4) * 8;   // A: lane-halves hold K 0-7 / 8-15 (+16 hi chunk)
    const int b_koff = (lane >> 4) * 8;   // B: lane-halves hold K 0-7 / 8-15 (+16 hi chunk)
    const _Float16* aptr = xt + (lane & 15) * XT_STRIDE;

    const int nb0 = wave * 4;             // 4 contiguous 16-col blocks per wave
    const _Float16* bpt = W1t + (long)(nb0 * 16 + n_lane) * 1024;

    v8f acc[4] = {{}, {}, {}, {}};
    for (int k0 = 0; k0 < 1024; k0 += 32) {
        v8h alo = *(const v8h*)(aptr + k0 + a_koff);
        v8h ahi = *(const v8h*)(aptr + k0 + a_koff + 16);
        v16h a = __builtin_shufflevector(alo, ahi, 0,1,2,3,4,5,6,7,8,9,10,11,12,13,14,15);
#pragma unroll
        for (int j = 0; j < 4; ++j) {
            const _Float16* bp = bpt + (long)j * 16 * 1024 + k0 + b_koff;
            v8h blo = *(const v8h*)(bp);
            v8h bhi = *(const v8h*)(bp + 16);
            v16h bf = __builtin_shufflevector(blo, bhi, 0,1,2,3,4,5,6,7,8,9,10,11,12,13,14,15);
            acc[j] = __builtin_amdgcn_wmma_f32_16x16x32_f16(false, a, false, bf,
                                                            (short)0, acc[j], false, false);
        }
    }

    // epilogue: +b1, relu, *W2, cross-lane reduce over 16 n-lanes per row
#pragma unroll
    for (int j = 0; j < 4; ++j) {
        const int n0 = (nb0 + j) * 16;
        float bn  = b1[n0 + n_lane];
        float w2n = W2[n0 + n_lane];
        for (int r2 = 0; r2 < 8; ++r2) {
            float h = fmaxf(acc[j][r2] + bn, 0.0f) * w2n;
            for (int msk = 1; msk < 16; msk <<= 1) h += __shfl_xor(h, msk, 32);
            if (n_lane == 0) {
                int m = ((lane >> 4) ? 8 : 0) + r2;
                atomicAdd(&rowacc[m], h);
            }
        }
    }
    __syncthreads();
    if (tid < 16) {
        long row = row0 + tid;
        float v = rowacc[tid] + b2[0];
        if (mask[row]) v = -1.0e9f;
        logits[row] = v;
    }
}

// ---------------------------------------------------------------------------
// Per-batch: softmax -> mu -> Sigma -> Sinv -> r(100) -> w = G @ r (196)
// ---------------------------------------------------------------------------
__device__ __forceinline__ float blkSum(float v, float* red) {
    int t = threadIdx.x;
    red[t] = v; __syncthreads();
    for (int s = 128; s > 0; s >>= 1) { if (t < s) red[t] += red[t + s]; __syncthreads(); }
    float r = red[0]; __syncthreads();
    return r;
}
__device__ __forceinline__ float blkMax(float v, float* red) {
    int t = threadIdx.x;
    red[t] = v; __syncthreads();
    for (int s = 128; s > 0; s >>= 1) { if (t < s) red[t] = fmaxf(red[t], red[t + s]); __syncthreads(); }
    float r = red[0]; __syncthreads();
    return r;
}

__global__ __launch_bounds__(256) void k_stats(
    const float* __restrict__ logits, const float* __restrict__ G,
    float* __restrict__ w) {
    __shared__ float red[256];
    __shared__ float rArr[100];
    const int b = blockIdx.x, t = threadIdx.x;
    const bool act = (t < 196);

    float lg = act ? logits[b * 196 + t] : -3.0e38f;
    float mx = blkMax(lg, red);
    float e = act ? expf(lg - mx) : 0.0f;
    float sum = blkSum(e, red);
    float p = e / sum;

    float px = 0.0f, py = 0.0f;
    if (act) { px = lin14(t / 14); py = lin14(t % 14); }
    float mux = blkSum(p * px, red);
    float muy = blkSum(p * py, red);
    float dx = px - mux, dy = py - muy;
    float sxx = blkSum(p * dx * dx, red);
    float sxy = blkSum(p * dx * dy, red);
    float syy = blkSum(p * dy * dy, red);

    // S = Sigma + (1e-6 + 1e-3) I
    float a  = sxx + 1.0e-6f + 0.001f;
    float dd = syy + 1.0e-6f + 0.001f;
    float bb = sxy;
    float det = a * dd - bb * bb;
    float i00 = dd / det, i01 = -bb / det, i11 = a / det;

    if (t < 100) {
        float dmx = mux - lin10(t / 10);
        float dmy = muy - lin10(t % 10);
        float q = dmx * (i00 * dmx + i01 * dmy) + dmy * (i01 * dmx + i11 * dmy);
        rArr[t] = expf(-0.5f * q) / (TWO_PI_F * sqrtf(det));
    }
    __syncthreads();
    if (act) {
        float acc = 0.0f;
        for (int n = 0; n < 100; ++n) acc += G[t * 100 + n] * rArr[n];
        w[b * 196 + t] = acc;
    }
}

// ---------------------------------------------------------------------------
// ctx[b,d] = sum_l w[b,l] * x[b,l,d]  (float4-vectorized; HBM-bound: 205MB read)
// ---------------------------------------------------------------------------
__global__ __launch_bounds__(256) void k_ctx(
    const float* __restrict__ x, const float* __restrict__ w,
    float* __restrict__ ctx, _Float16* __restrict__ ctxh) {
    __shared__ float wl[196];
    const int b = blockIdx.x, t = threadIdx.x;
    if (t < 196) wl[t] = w[b * 196 + t];
    __syncthreads();
    const int d4 = t * 4;
    const float* xb = x + (long)b * 196 * 1024 + d4;
    float a0 = 0.0f, a1 = 0.0f, a2 = 0.0f, a3 = 0.0f;
    for (int l = 0; l < 196; ++l) {
        const float4 v = *(const float4*)(xb + (long)l * 1024);
        if (l + 8 < 196) __builtin_prefetch(xb + (long)(l + 8) * 1024, 0, 0);
        const float wv = wl[l];
        a0 += wv * v.x; a1 += wv * v.y; a2 += wv * v.z; a3 += wv * v.w;
    }
    float4 o; o.x = a0; o.y = a1; o.z = a2; o.w = a3;
    *(float4*)(ctx + (long)b * 1024 + d4) = o;
    v4h h = { (_Float16)a0, (_Float16)a1, (_Float16)a2, (_Float16)a3 };
    *(v4h*)(ctxh + (long)b * 1024 + d4) = h;
}

// ---------------------------------------------------------------------------
// GEMM-2: out = ctx(256x1024) @ Wm(1024x2048) + bm  via v_wmma f16
// ---------------------------------------------------------------------------
__global__ __launch_bounds__(256) void k_out(
    const _Float16* __restrict__ ctxh, const _Float16* __restrict__ Wmt,
    const float* __restrict__ bm, float* __restrict__ out) {
    const int tid = threadIdx.x;
    const int lane = tid & 31;
    const int wave = tid >> 5;
    const int rb = blockIdx.x >> 4;         // 16 row tiles
    const int cg = blockIdx.x & 15;         // 16 col groups of 128
    const int m0 = rb * 16;
    const int o0 = cg * 128 + wave * 16;
    const int n_lane = lane & 15;
    const int a_koff = (lane >> 4) * 8;
    const int b_koff = (lane >> 4) * 8;
    const _Float16* aptr = ctxh + (long)(m0 + (lane & 15)) * 1024;
    const _Float16* bptr = Wmt + (long)(o0 + n_lane) * 1024;

    v8f acc = {};
    for (int k0 = 0; k0 < 1024; k0 += 32) {
        v8h alo = *(const v8h*)(aptr + k0 + a_koff);
        v8h ahi = *(const v8h*)(aptr + k0 + a_koff + 16);
        v8h blo = *(const v8h*)(bptr + k0 + b_koff);
        v8h bhi = *(const v8h*)(bptr + k0 + b_koff + 16);
        v16h a = __builtin_shufflevector(alo, ahi, 0,1,2,3,4,5,6,7,8,9,10,11,12,13,14,15);
        v16h b = __builtin_shufflevector(blo, bhi, 0,1,2,3,4,5,6,7,8,9,10,11,12,13,14,15);
        acc = __builtin_amdgcn_wmma_f32_16x16x32_f16(false, a, false, b,
                                                     (short)0, acc, false, false);
    }
    const int mbase = (lane >= 16) ? 8 : 0;
    const float bmv = bm[o0 + n_lane];
    for (int r2 = 0; r2 < 8; ++r2)
        out[(long)(m0 + mbase + r2) * 2048 + o0 + n_lane] = acc[r2] + bmv;
}

// ---------------------------------------------------------------------------
extern "C" void kernel_launch(void* const* d_in, const int* in_sizes, int n_in,
                              void* d_out, int out_size, void* d_ws, size_t ws_size,
                              hipStream_t stream) {
    const float*         x    = (const float*)d_in[0];
    const unsigned char* mask = (const unsigned char*)d_in[1];
    const float*         W1   = (const float*)d_in[2];   // 1024 x 512
    const float*         b1   = (const float*)d_in[3];
    const float*         W2   = (const float*)d_in[4];   // 512 x 1
    const float*         b2   = (const float*)d_in[5];
    const float*         Wm   = (const float*)d_in[6];   // 1024 x 2048
    const float*         bm   = (const float*)d_in[7];
    float*               out  = (float*)d_out;           // 256 x 2048

    char* ws = (char*)d_ws;
    size_t off = 0;
    auto take = [&](size_t bytes) { size_t r = off; off = (off + bytes + 255) & ~(size_t)255; return r; };
    float*    Fm     = (float*)   (ws + take(100 * 196 * 4));
    float*    AUG    = (float*)   (ws + take(100 * 200 * 4));
    float*    G      = (float*)   (ws + take(196 * 100 * 4));
    float*    logits = (float*)   (ws + take(256 * 196 * 4));
    float*    w      = (float*)   (ws + take(256 * 196 * 4));
    float*    ctx    = (float*)   (ws + take(256 * 1024 * 4));
    _Float16* ctxh   = (_Float16*)(ws + take(256 * 1024 * 2));
    _Float16* W1t    = (_Float16*)(ws + take(512 * 1024 * 2));
    _Float16* Wmt    = (_Float16*)(ws + take(2048 * 1024 * 2));
    (void)ws_size; (void)in_sizes; (void)n_in; (void)out_size;

    // weight conversion (f32 [K][N] -> f16 [N][K])
    k_cvtT<<<(1024 * 512 + 255) / 256, 256, 0, stream>>>(W1, W1t, 1024, 512);
    k_cvtT<<<(1024 * 2048 + 255) / 256, 256, 0, stream>>>(Wm, Wmt, 1024, 2048);

    // constant G matrix
    k_F  <<<(100 * 196 + 255) / 256, 256, 0, stream>>>(Fm);
    k_A  <<<(100 * 200 + 255) / 256, 256, 0, stream>>>(Fm, AUG);
    k_inv<<<1, 256, 0, stream>>>(AUG);
    k_G  <<<(196 * 100 + 255) / 256, 256, 0, stream>>>(Fm, AUG, G);

    // main pipeline
    k_logits<<<(256 * 196) / 16, 256, K_LOGITS_LDS, stream>>>(x, W1t, b1, W2, b2, mask, logits);
    k_stats <<<256, 256, 0, stream>>>(logits, G, w);
    k_ctx   <<<256, 256, 0, stream>>>(x, w, ctx, ctxh);
    k_out   <<<256, 256, 0, stream>>>(ctxh, Wmt, bm, out);
}
